// GCN_67542655696999
// MI455X (gfx1250) — matrix operand order repaired and checked by reading, not compiled, not run on hardware
//
#include <hip/hip_runtime.h>

typedef __attribute__((ext_vector_type(16))) __bf16 v16bf;
typedef __attribute__((ext_vector_type(8)))  __bf16 v8bf;
typedef __attribute__((ext_vector_type(4)))  __bf16 v4bf;
typedef __attribute__((ext_vector_type(8)))  float  v8f;

#define D_IN   256
#define D_HID  128
#define N_CLS  47
#define N_CLSP 48   // padded to 3 WMMA col-tiles

// fp32 -> bf16, round-to-nearest-even (bit-exact, no reliance on fptrunc)
__device__ __forceinline__ __bf16 f32_to_bf16(float f) {
  unsigned u = __builtin_bit_cast(unsigned, f);
  u += 0x7FFFu + ((u >> 16) & 1u);
  unsigned short h = (unsigned short)(u >> 16);
  return __builtin_bit_cast(__bf16, h);
}

// hardware fp32 atomic add (global_atomic_add_f32) — resolved at L2 atomic units
__device__ __forceinline__ void atomAddF(float* p, float v) {
  unsafeAtomicAdd(p, v);
}

// ---------------- degree / symmetric norm ----------------

__global__ void k_deg_init(float* deg, int n) {
  int i = blockIdx.x * blockDim.x + threadIdx.x;
  if (i < n) deg[i] = 1.0f;  // self-loop contributes 1
}

__global__ void k_deg_edges(const long long* __restrict__ ei, float* deg, long long E) {
  long long e = (long long)blockIdx.x * blockDim.x + threadIdx.x;
  if (e >= E) return;
  int d = (int)ei[E + e];
  atomAddF(&deg[d], 1.0f);
}

__global__ void k_rsqrt_inplace(float* deg, int n) {
  int i = blockIdx.x * blockDim.x + threadIdx.x;
  if (i < n) deg[i] = rsqrtf(deg[i]);  // deg >= 1 always
}

// ---------------- operand preparation ----------------

// Xb = bf16(X), packed 4-wide (float4 load -> v4bf store)
__global__ void k_cvt_x(const float* __restrict__ X, __bf16* __restrict__ Xb, long long total4) {
  long long t = (long long)blockIdx.x * blockDim.x + threadIdx.x;
  if (t >= total4) return;
  float4 v = ((const float4*)X)[t];
  v4bf o = { f32_to_bf16(v.x), f32_to_bf16(v.y), f32_to_bf16(v.z), f32_to_bf16(v.w) };
  ((v4bf*)Xb)[t] = o;
}

// Pack W1 [256,128] f32 into B-fragment order:
// frag = (ntile*8 + ktile), 32 lanes x 16 halfs contiguous per lane.
// slot e of lane l -> K = ktile*32 + (l>>4)*8 + (e&7) + (e>>3)*16, col = ntile*16 + (l&15)
__global__ void k_packw1(const float* __restrict__ W1, __bf16* __restrict__ W1p) {
  int t = blockIdx.x * blockDim.x + threadIdx.x;
  if (t >= 8 * 8 * 32 * 16) return;                 // 32768
  int e = t & 15, lane = (t >> 4) & 31, kt = (t >> 9) & 7, nt = t >> 12;
  int k = kt * 32 + (lane >> 4) * 8 + (e & 7) + (e >> 3) * 16;
  int c = nt * 16 + (lane & 15);
  W1p[t] = f32_to_bf16(W1[(size_t)k * D_HID + c]);
}

// Pack W2 [128,47] f32 into B-fragment order, zero-padded to 48 cols.
__global__ void k_packw2(const float* __restrict__ W2, __bf16* __restrict__ W2p) {
  int t = blockIdx.x * blockDim.x + threadIdx.x;
  if (t >= 3 * 4 * 32 * 16) return;                 // 6144
  int e = t & 15, lane = (t >> 4) & 31, kt = (t >> 9) & 3, nt = t >> 11;
  int k = kt * 32 + (lane >> 4) * 8 + (e & 7) + (e >> 3) * 16;
  int c = nt * 16 + (lane & 15);
  W2p[t] = (c < N_CLS) ? f32_to_bf16(W2[(size_t)k * N_CLS + c]) : f32_to_bf16(0.0f);
}

// ---------------- WMMA GEMM 1: H[N,128] = Xb[N,256] @ W1p ----------------
// One wave32 per 16x16 f32 tile. A frag: row-major bf16 -> two b128 loads per K-tile
// (slots 0-7: K = half*8+0..7 ; slots 8-15: K = 16+half*8+0..7).
// B frag: pre-packed, one contiguous 32B load per lane. C/D: col = l&15, row = v + 8*half.

__global__ __launch_bounds__(256) void k_gemm1(const __bf16* __restrict__ Xb,
                                               const __bf16* __restrict__ W1p,
                                               float* __restrict__ H, int M) {
  const int lane = threadIdx.x & 31;
  const int wave = threadIdx.x >> 5;
  const int tile = blockIdx.x * 8 + wave;
  const int ntile = tile & 7;          // 128/16 = 8 col tiles
  const int mtile = tile >> 3;
  if (mtile * 16 >= M) return;
  const int half = lane >> 4;
  const int r    = lane & 15;
  const int m0 = mtile * 16;
  const int n0 = ntile * 16;

  const __bf16* xrow  = Xb + (size_t)(m0 + r) * D_IN + half * 8;
  const v16bf*  bfrag = (const v16bf*)W1p + (size_t)ntile * 8 * 32 + lane;

  v8f acc = {};
#pragma unroll
  for (int kt = 0; kt < 8; ++kt) {                  // K = 256 = 8 x 32
    v8bf alo = *(const v8bf*)(xrow + kt * 32);
    v8bf ahi = *(const v8bf*)(xrow + kt * 32 + 16);
    v16bf a = __builtin_shufflevector(alo, ahi, 0, 1, 2, 3, 4, 5, 6, 7,
                                      8, 9, 10, 11, 12, 13, 14, 15);
    v16bf b = bfrag[(size_t)kt * 32];
    acc = __builtin_amdgcn_wmma_f32_16x16x32_bf16(false, a, false, b,
                                                  (short)0, acc, false, false);
  }

  float* hrow = H + (size_t)m0 * D_HID + n0 + r;
#pragma unroll
  for (int v = 0; v < 8; ++v)
    hrow[(size_t)(v + half * 8) * D_HID] = acc[v];
}

// ---------------- layer-1 aggregation ----------------

// agg[i] = H[i] * dinv[i]^2  (self-loop term; also initializes agg)
__global__ void k_selfinit1(const float* __restrict__ H, const float* __restrict__ dinv,
                            float* __restrict__ agg, int n) {
  int t = blockIdx.x * blockDim.x + threadIdx.x;   // n*32 threads, 4 feats each
  if (t >= n * 32) return;
  int i = t >> 5;
  int c = (t & 31) * 4;
  float s = dinv[i]; s = s * s;
  float4 v = *(const float4*)(H + (size_t)i * D_HID + c);
  float4* o = (float4*)(agg + (size_t)i * D_HID + c);
  *o = make_float4(v.x * s, v.y * s, v.z * s, v.w * s);
}

__global__ void k_edge1(const long long* __restrict__ ei, const float* __restrict__ H,
                        const float* __restrict__ dinv, float* __restrict__ agg, long long E) {
  long long t = (long long)blockIdx.x * blockDim.x + threadIdx.x;  // E*32 threads
  if (t >= E * 32) return;
  long long e = t >> 5;
  int c = (int)(t & 31) * 4;
  int s = (int)ei[e];
  int d = (int)ei[E + e];
  float nrm = dinv[s] * dinv[d];
  float4 v = *(const float4*)(H + (size_t)s * D_HID + c);
  float* o = agg + (size_t)d * D_HID + c;
  atomAddF(o + 0, v.x * nrm);
  atomAddF(o + 1, v.y * nrm);
  atomAddF(o + 2, v.z * nrm);
  atomAddF(o + 3, v.w * nrm);
}

// h1b = bf16(relu(agg + b1)), packed 4-wide
__global__ void k_relu_cvt(const float* __restrict__ agg, const float* __restrict__ b1,
                           __bf16* __restrict__ h1b, int total4) {
  int t = blockIdx.x * blockDim.x + threadIdx.x;   // total4 = N*128/4
  if (t >= total4) return;
  int c = (t & (D_HID / 4 - 1)) * 4;
  float4 v = ((const float4*)agg)[t];
  float4 bb = *(const float4*)(b1 + c);
  v4bf o = { f32_to_bf16(fmaxf(v.x + bb.x, 0.0f)),
             f32_to_bf16(fmaxf(v.y + bb.y, 0.0f)),
             f32_to_bf16(fmaxf(v.z + bb.z, 0.0f)),
             f32_to_bf16(fmaxf(v.w + bb.w, 0.0f)) };
  ((v4bf*)h1b)[t] = o;
}

// ---------------- WMMA GEMM 2: H2[N,48] = h1b[N,128] @ W2p ----------------

__global__ __launch_bounds__(256) void k_gemm2(const __bf16* __restrict__ A,
                                               const __bf16* __restrict__ W2p,
                                               float* __restrict__ H2, int M) {
  const int lane = threadIdx.x & 31;
  const int wave = threadIdx.x >> 5;
  const int tile = blockIdx.x * 8 + wave;
  const int NT = N_CLSP / 16;          // 3 col tiles
  const int ntile = tile % NT;
  const int mtile = tile / NT;
  if (mtile * 16 >= M) return;
  const int half = lane >> 4;
  const int r    = lane & 15;
  const int m0 = mtile * 16;
  const int n0 = ntile * 16;

  const __bf16* arow  = A + (size_t)(m0 + r) * D_HID + half * 8;
  const v16bf*  bfrag = (const v16bf*)W2p + (size_t)ntile * 4 * 32 + lane;

  v8f acc = {};
#pragma unroll
  for (int kt = 0; kt < 4; ++kt) {                  // K = 128 = 4 x 32
    v8bf alo = *(const v8bf*)(arow + kt * 32);
    v8bf ahi = *(const v8bf*)(arow + kt * 32 + 16);
    v16bf a = __builtin_shufflevector(alo, ahi, 0, 1, 2, 3, 4, 5, 6, 7,
                                      8, 9, 10, 11, 12, 13, 14, 15);
    v16bf b = bfrag[(size_t)kt * 32];
    acc = __builtin_amdgcn_wmma_f32_16x16x32_bf16(false, a, false, b,
                                                  (short)0, acc, false, false);
  }

  float* hrow = H2 + (size_t)m0 * N_CLSP + n0 + r;
#pragma unroll
  for (int v = 0; v < 8; ++v)
    hrow[(size_t)(v + half * 8) * N_CLSP] = acc[v];
}

// ---------------- layer-2 aggregation ----------------

// out[i,c] = b2[c] + H2[i,c] * dinv[i]^2  (self-loop; also initializes out)
__global__ void k_selfinit2(const float* __restrict__ H2, const float* __restrict__ dinv,
                            const float* __restrict__ b2, float* __restrict__ out, int n) {
  int t = blockIdx.x * blockDim.x + threadIdx.x;   // n*48
  if (t >= n * N_CLSP) return;
  int i = t / N_CLSP, c = t % N_CLSP;
  if (c >= N_CLS) return;
  float s = dinv[i]; s = s * s;
  out[(size_t)i * N_CLS + c] = b2[c] + H2[(size_t)i * N_CLSP + c] * s;
}

__global__ void k_edge2(const long long* __restrict__ ei, const float* __restrict__ H2,
                        const float* __restrict__ dinv, float* __restrict__ out, long long E) {
  long long t = (long long)blockIdx.x * blockDim.x + threadIdx.x;  // E*48
  if (t >= E * N_CLSP) return;
  long long e = t / N_CLSP;
  int c = (int)(t % N_CLSP);
  if (c >= N_CLS) return;
  int s = (int)ei[e];
  int d = (int)ei[E + e];
  float nrm = dinv[s] * dinv[d];
  atomAddF(&out[(size_t)d * N_CLS + c], H2[(size_t)s * N_CLSP + c] * nrm);
}

// ---------------- launcher ----------------

extern "C" void kernel_launch(void* const* d_in, const int* in_sizes, int n_in,
                              void* d_out, int out_size, void* d_ws, size_t ws_size,
                              hipStream_t stream) {
  const float*     x  = (const float*)d_in[0];
  const long long* ei = (const long long*)d_in[1];
  const float*     W1 = (const float*)d_in[2];
  const float*     b1 = (const float*)d_in[3];
  const float*     W2 = (const float*)d_in[4];
  const float*     b2 = (const float*)d_in[5];
  float* out = (float*)d_out;

  const int       N = in_sizes[0] / D_IN;     // 100000
  const long long E = in_sizes[1] / 2;        // 1600000

  // workspace layout (256B aligned slabs, with lifetime-based aliasing)
  char* ws = (char*)d_ws;
  size_t off = 0;
  auto alloc = [&](size_t bytes) -> char* {
    char* p = ws + off;
    off += (bytes + 255) & ~(size_t)255;
    return p;
  };
  float*  dinv = (float*)alloc((size_t)N * 4);            // deg -> dinv in place
  __bf16* xb   = (__bf16*)alloc((size_t)N * D_IN * 2);    // 51.2 MB; dead after gemm1
  float*  H    = (float*)alloc((size_t)N * D_HID * 4);    // 51.2 MB; dead after relu_cvt
  __bf16* h1b  = (__bf16*)alloc((size_t)N * D_HID * 2);
  __bf16* w1p  = (__bf16*)alloc((size_t)8 * 8 * 32 * 16 * 2);
  __bf16* w2p  = (__bf16*)alloc((size_t)3 * 4 * 32 * 16 * 2);
  float*  agg  = (float*)xb;   // alias: same 51.2 MB footprint, disjoint lifetime
  float*  H2   = H;            // alias: H dead once h1b exists

  const int B256 = 256;
  const int mtiles = (N + 15) / 16;

  // degrees / symmetric norm
  k_deg_init<<<(N + B256 - 1) / B256, B256, 0, stream>>>(dinv, N);
  k_deg_edges<<<(unsigned)((E + B256 - 1) / B256), B256, 0, stream>>>(ei, dinv, E);
  k_rsqrt_inplace<<<(N + B256 - 1) / B256, B256, 0, stream>>>(dinv, N);

  // operand prep
  k_cvt_x<<<(unsigned)(((long long)N * D_IN / 4 + B256 - 1) / B256), B256, 0, stream>>>(
      x, xb, (long long)N * D_IN / 4);
  k_packw1<<<(8 * 8 * 32 * 16 + B256 - 1) / B256, B256, 0, stream>>>(W1, w1p);
  k_packw2<<<(3 * 4 * 32 * 16 + B256 - 1) / B256, B256, 0, stream>>>(W2, w2p);

  // layer 1
  k_gemm1<<<mtiles, B256, 0, stream>>>(xb, w1p, H, N);    // mtiles*8 waves = all tiles
  k_selfinit1<<<((size_t)N * 32 + B256 - 1) / B256, B256, 0, stream>>>(H, dinv, agg, N);
  k_edge1<<<(unsigned)((E * 32 + B256 - 1) / B256), B256, 0, stream>>>(ei, H, dinv, agg, E);
  k_relu_cvt<<<((size_t)N * D_HID / 4 + B256 - 1) / B256, B256, 0, stream>>>(
      agg, b1, h1b, N * D_HID / 4);

  // layer 2
  k_gemm2<<<(mtiles * 3 + 7) / 8, B256, 0, stream>>>(h1b, w2p, H2, N);
  k_selfinit2<<<((size_t)N * N_CLSP + B256 - 1) / B256, B256, 0, stream>>>(H2, dinv, b2, out, N);
  k_edge2<<<(unsigned)((E * N_CLSP + B256 - 1) / B256), B256, 0, stream>>>(ei, H2, dinv, out, E);
}
